// MatchAttention_57930518889207
// MI455X (gfx1250) — matrix operand order repaired and checked by reading, not compiled, not run on hardware
//
#include <hip/hip_runtime.h>

// Problem dims
#define B_  32
#define L1_ 256
#define L2_ 64
#define D1_ 256
#define D2_ 256
#define H_  256
#define P_  128
#define I_  512   // D1 + D2

typedef __bf16 bf16_t;
typedef __attribute__((ext_vector_type(16))) __bf16 v16bf;
typedef __attribute__((ext_vector_type(8)))  float  v8f;
typedef __attribute__((ext_vector_type(4)))  float  vf4;

static __device__ __forceinline__ v8f v8f_zero() {
  v8f z = {0.f, 0.f, 0.f, 0.f, 0.f, 0.f, 0.f, 0.f};
  return z;
}

static __device__ __forceinline__ v8f wmma_bf16(v16bf a, v16bf b, v8f c) {
  // D(16x16,f32) = A(16x32,bf16) x B(32x16,bf16) + C
  return __builtin_amdgcn_wmma_f32_16x16x32_bf16(
      /*neg_a=*/false, a, /*neg_b=*/false, b,
      /*c_mod=*/(short)0, c, /*reuse_a=*/false, /*reuse_b=*/false);
}

// A fragment: 16x32 bf16 tile, row-major source with row stride `rs` (elements).
// ISA layout: lanes 0-15 -> row=lane, K = {k0..k0+7, k0+16..k0+23}
//             lanes16-31 -> row=lane-15? no: row=lane&15, K = {k0+8..15, k0+24..31}
static __device__ __forceinline__ v16bf load_frag_a(const bf16_t* base, int rs,
                                                    int k0, int lane) {
  int row  = lane & 15;
  int koff = k0 + ((lane & 16) ? 8 : 0);
  const bf16_t* p = base + (size_t)row * rs + koff;
  union { v16bf v; vf4 q[2]; } u;
  u.q[0] = *(const vf4*)(p);        // 8 bf16: K = koff .. koff+7
  u.q[1] = *(const vf4*)(p + 16);   // 8 bf16: K = koff+16 .. koff+23
  return u.v;
}

// B fragment: 32x16 bf16 tile from weight matrix stored [N][K] row-major
// (i.e. out = act @ W^T with W[n][k] contiguous in k -> B[k][n] = W[n][k]).
// Layout: lanes 0-15 -> col n0+lane, K = {k0..k0+7, k0+16..k0+23}; lanes 16-31 -> K += 8.
static __device__ __forceinline__ v16bf load_frag_b(const bf16_t* W, int Ktot,
                                                    int n0, int k0, int lane) {
  int n    = n0 + (lane & 15);
  int koff = k0 + ((lane & 16) ? 8 : 0);
  const bf16_t* p = W + (size_t)n * Ktot + koff;
  union { v16bf v; vf4 q[2]; } u;
  u.q[0] = *(const vf4*)(p);        // K = koff .. koff+7
  u.q[1] = *(const vf4*)(p + 16);   // K = koff+16 .. koff+23
  return u.v;
}

static __device__ __forceinline__ float tanh_fast(float x) {
  // stable: e->inf => 1, e->0 => -1
  float e = __expf(2.f * x);
  return 1.f - 2.f / (e + 1.f);
}
static __device__ __forceinline__ float sigmoid_fast(float x) {
  return 1.f / (1.f + __expf(-x));
}

// ---------------- precompute kernels ----------------

__global__ void cvt_bf16_kernel(bf16_t* __restrict__ dst,
                                const float* __restrict__ src, int n) {
  for (int i = blockIdx.x * blockDim.x + threadIdx.x; i < n;
       i += gridDim.x * blockDim.x)
    dst[i] = (bf16_t)src[i];
}

// dst[c*R + r] = src[r*C + c]  (produce [N][K]-major bf16 from [K][N] fp32)
__global__ void tcvt_bf16_kernel(bf16_t* __restrict__ dst,
                                 const float* __restrict__ src, int R, int C) {
  int n = R * C;
  for (int i = blockIdx.x * blockDim.x + threadIdx.x; i < n;
       i += gridDim.x * blockDim.x) {
    int r = i / C, c = i % C;
    dst[(size_t)c * R + r] = (bf16_t)src[i];
  }
}

// seq2_proj[m][p] = sum_k seq2bf[m][k] * W1t[p][k];  m in [0, B*L2), K=D2, N=P
__global__ __launch_bounds__(256, 1)
void seq2proj_kernel(float* __restrict__ proj, const bf16_t* __restrict__ s2bf,
                     const bf16_t* __restrict__ W1t) {
  int lane = threadIdx.x & 31;
  int wave = threadIdx.x >> 5;           // 8 waves -> 8 N-tiles (P=128)
  int m0   = blockIdx.x * 16;            // 128 blocks
  int n0   = wave * 16;
  const bf16_t* A = s2bf + (size_t)m0 * D2_;
  v8f c = v8f_zero();
#pragma unroll
  for (int k0 = 0; k0 < D2_; k0 += 32)
    c = wmma_bf16(load_frag_a(A, D2_, k0, lane),
                  load_frag_b(W1t, D2_, n0, k0, lane), c);
  int n = lane & 15;
#pragma unroll
  for (int r = 0; r < 8; ++r) {
    int m = r + ((lane & 16) ? 8 : 0);
    proj[(size_t)(m0 + m) * P_ + n0 + n] = c[r];
  }
}

// ---------------- persistent recurrence kernel ----------------
// grid = 2 workgroups (16 batch rows each), 512 threads = 16 waves.

__global__ __launch_bounds__(512, 1)
void matchatt_persistent(const float* __restrict__ seq1,
                         const float* __restrict__ seq2,
                         const float* __restrict__ mask1,
                         const float* __restrict__ mask2,
                         const float* __restrict__ vattn,
                         const float* __restrict__ bg,
                         const float* __restrict__ bih,
                         const float* __restrict__ bhh,
                         const bf16_t* __restrict__ Wg,    // [512][512] bf16
                         const bf16_t* __restrict__ wih,   // [768][512] bf16
                         const bf16_t* __restrict__ whh,   // [768][256] bf16
                         const bf16_t* __restrict__ W2t,   // [128][512] bf16
                         const float* __restrict__ s2proj, // [B*L2][128] f32
                         float* __restrict__ out) {
  extern __shared__ char smem[];
  float*  h32    = (float*)smem;                       // [16][256] 16KB
  float*  proj   = h32    + 16 * H_;                   // [16][128]  8KB
  float*  scores = proj   + 16 * P_;                   // [16][64]   4KB
  bf16_t* vecA   = (bf16_t*)(scores + 16 * L2_);       // [16][512] 16KB  [x;h]
  bf16_t* inpA   = vecA   + 16 * I_;                   // [16][512] 16KB  [x;att]
  bf16_t* gateA  = inpA   + 16 * I_;                   // [16][512] 16KB  gated

  const int tid   = threadIdx.x;
  const int lane  = tid & 31;
  const int wave  = tid >> 5;            // 0..15
  const int bbase = blockIdx.x * 16;     // batch rows handled by this WG

  // init h = 0 (fp32 and bf16 halves)
  for (int i = tid; i < 16 * H_; i += 512) {
    h32[i] = 0.f;
    vecA[(i >> 8) * I_ + D1_ + (i & 255)] = (bf16_t)0.f;
  }
  __syncthreads();

  for (int t = 0; t < L1_; ++t) {
    // ---- phase 0: stage x_t into vecA[:,0:256] and inpA[:,0:256] (bf16) ----
    for (int i = tid; i < 16 * D1_; i += 512) {
      int b = i >> 8, d = i & 255;
      float x = seq1[((size_t)(bbase + b) * L1_ + t) * D1_ + d];
      bf16_t xb = (bf16_t)x;
      vecA[b * I_ + d] = xb;
      inpA[b * I_ + d] = xb;
    }
    __syncthreads();

    // ---- phase 1: proj = [x;h] @ W2  (M=16, N=128, K=512), waves 0-7 ----
    if (wave < 8) {
      int n0 = wave * 16;
      v8f c = v8f_zero();
#pragma unroll
      for (int k0 = 0; k0 < I_; k0 += 32)
        c = wmma_bf16(load_frag_a(vecA, I_, k0, lane),
                      load_frag_b(W2t, I_, n0, k0, lane), c);
      int n = lane & 15;
#pragma unroll
      for (int r = 0; r < 8; ++r) {
        int m = r + ((lane & 16) ? 8 : 0);
        proj[m * P_ + n0 + n] = c[r];
      }
    }
    __syncthreads();

    // ---- phase 2: scores s[b][l] = v . tanh(s2proj[b,l,:] + proj[b,:]) ----
    for (int i = tid; i < 16 * L2_; i += 512) {
      int b = i >> 6, l = i & 63;
      const float* sp = s2proj + ((size_t)(bbase + b) * L2_ + l) * P_;
      const float* pr = proj + b * P_;
      float acc = 0.f;
#pragma unroll 4
      for (int p = 0; p < P_; ++p)
        acc += vattn[p] * tanh_fast(sp[p] + pr[p]);
      float m2 = mask2[(size_t)(bbase + b) * L2_ + l];
      scores[b * L2_ + l] = acc + (m2 > 0.f ? 0.f : -1e9f);
    }
    __syncthreads();

    // ---- phase 3: softmax over L2=64, one wave per batch row ----
    {
      int b = wave;
      float s0 = scores[b * L2_ + lane];
      float s1 = scores[b * L2_ + 32 + lane];
      float mx = fmaxf(s0, s1);
      for (int off = 16; off; off >>= 1) mx = fmaxf(mx, __shfl_xor(mx, off, 32));
      float e0 = __expf(s0 - mx), e1 = __expf(s1 - mx);
      float sm = e0 + e1;
      for (int off = 16; off; off >>= 1) sm += __shfl_xor(sm, off, 32);
      float inv = 1.f / sm;
      scores[b * L2_ + lane]      = e0 * inv;
      scores[b * L2_ + 32 + lane] = e1 * inv;
    }
    __syncthreads();

    // ---- phase 4: att[b][d] = sum_l w[b][l]*seq2[b][l][d] -> inpA[:,256:] ----
    for (int i = tid; i < 16 * D2_; i += 512) {
      int b = i >> 8, d = i & 255;
      const float* s2 = seq2 + ((size_t)(bbase + b) * L2_) * D2_ + d;
      const float* w  = scores + b * L2_;
      float acc = 0.f;
#pragma unroll 8
      for (int l = 0; l < L2_; ++l) acc += w[l] * s2[(size_t)l * D2_];
      inpA[b * I_ + D1_ + d] = (bf16_t)acc;
    }
    __syncthreads();

    // ---- phase 5: gated = sigmoid(inp @ Wg^T + bg) * inp (N=512, 2 tiles/wave)
#pragma unroll
    for (int j = 0; j < 2; ++j) {
      int n0 = (wave * 2 + j) * 16;
      v8f c = v8f_zero();
#pragma unroll
      for (int k0 = 0; k0 < I_; k0 += 32) {
        __builtin_prefetch((const void*)(Wg + (size_t)(n0 + (lane & 15)) * I_ +
                                         k0 + 64), 0, 0);
        c = wmma_bf16(load_frag_a(inpA, I_, k0, lane),
                      load_frag_b(Wg, I_, n0, k0, lane), c);
      }
      int n = lane & 15;
      float bgv = bg[n0 + n];
#pragma unroll
      for (int r = 0; r < 8; ++r) {
        int m = r + ((lane & 16) ? 8 : 0);
        float g  = sigmoid_fast(c[r] + bgv);
        float iv = (float)inpA[m * I_ + n0 + n];
        gateA[m * I_ + n0 + n] = (bf16_t)(g * iv);
      }
    }
    __syncthreads();

    // ---- phase 6: fused GRU. wave -> H-tile n0=wave*16; 6 accumulators ----
    {
      int n0 = wave * 16;
      v8f cir = v8f_zero(), ciz = v8f_zero(), cin = v8f_zero();
      v8f chr_ = v8f_zero(), chz = v8f_zero(), chn = v8f_zero();
#pragma unroll
      for (int k0 = 0; k0 < I_; k0 += 32) {   // gi = gated @ w_ih^T (K=512)
        __builtin_prefetch((const void*)(wih + (size_t)(n0 + (lane & 15)) * I_ +
                                         k0 + 64), 0, 0);
        v16bf a = load_frag_a(gateA, I_, k0, lane);
        cir = wmma_bf16(a, load_frag_b(wih, I_, n0,            k0, lane), cir);
        ciz = wmma_bf16(a, load_frag_b(wih, I_, H_ + n0,       k0, lane), ciz);
        cin = wmma_bf16(a, load_frag_b(wih, I_, 2 * H_ + n0,   k0, lane), cin);
      }
#pragma unroll
      for (int k0 = 0; k0 < H_; k0 += 32) {   // gh = h @ w_hh^T (K=256)
        v16bf a = load_frag_a(vecA + D1_, I_, k0, lane);  // h (bf16) columns
        chr_ = wmma_bf16(a, load_frag_b(whh, H_, n0,          k0, lane), chr_);
        chz  = wmma_bf16(a, load_frag_b(whh, H_, H_ + n0,     k0, lane), chz);
        chn  = wmma_bf16(a, load_frag_b(whh, H_, 2 * H_ + n0, k0, lane), chn);
      }
      // all waves finished reading h/gateA before anyone rewrites h
      __syncthreads();

      int n = lane & 15;
      float bir = bih[n0 + n], biz = bih[H_ + n0 + n], bin = bih[2 * H_ + n0 + n];
      float bhr = bhh[n0 + n], bhz = bhh[H_ + n0 + n], bhn = bhh[2 * H_ + n0 + n];
#pragma unroll
      for (int r = 0; r < 8; ++r) {
        int m = r + ((lane & 16) ? 8 : 0);
        float rg = sigmoid_fast((cir[r] + bir) + (chr_[r] + bhr));
        float zg = sigmoid_fast((ciz[r] + biz) + (chz[r] + bhz));
        float ng = tanh_fast((cin[r] + bin) + rg * (chn[r] + bhn));
        float ho = h32[m * H_ + n0 + n];
        float hn = (1.f - zg) * ng + zg * ho;
        h32[m * H_ + n0 + n] = hn;
        vecA[m * I_ + D1_ + n0 + n] = (bf16_t)hn;
        float mk = mask1[(size_t)(bbase + m) * L1_ + t];
        out[((size_t)(bbase + m) * L1_ + t) * H_ + n0 + n] = hn * mk;
      }
    }
    __syncthreads();
  }
}

// ---------------- host launcher ----------------

extern "C" void kernel_launch(void* const* d_in, const int* in_sizes, int n_in,
                              void* d_out, int out_size, void* d_ws,
                              size_t ws_size, hipStream_t stream) {
  (void)in_sizes; (void)n_in; (void)out_size; (void)ws_size;
  const float* seq1  = (const float*)d_in[0];
  const float* seq2  = (const float*)d_in[1];
  const float* mask1 = (const float*)d_in[2];
  const float* mask2 = (const float*)d_in[3];
  const float* W1    = (const float*)d_in[4];   // [256][128]
  const float* W2    = (const float*)d_in[5];   // [512][128]
  const float* vattn = (const float*)d_in[6];   // [128]
  const float* Wg    = (const float*)d_in[7];   // [512][512]
  const float* bg    = (const float*)d_in[8];
  const float* wih   = (const float*)d_in[9];   // [768][512]
  const float* whh   = (const float*)d_in[10];  // [768][256]
  const float* bih   = (const float*)d_in[11];
  const float* bhh   = (const float*)d_in[12];
  float* out = (float*)d_out;

  // workspace carve
  char* p = (char*)d_ws;
  bf16_t* wsWg  = (bf16_t*)p; p += (size_t)512 * 512 * 2;
  bf16_t* wswih = (bf16_t*)p; p += (size_t)768 * 512 * 2;
  bf16_t* wswhh = (bf16_t*)p; p += (size_t)768 * 256 * 2;
  bf16_t* wsW2t = (bf16_t*)p; p += (size_t)128 * 512 * 2;
  bf16_t* wsW1t = (bf16_t*)p; p += (size_t)128 * 256 * 2;
  bf16_t* wss2  = (bf16_t*)p; p += (size_t)B_ * L2_ * D2_ * 2;
  float*  wsprj = (float*)p;  p += (size_t)B_ * L2_ * P_ * 4;

  // weight / input conversion (stay [N][K]-major; W2,W1 need transpose)
  cvt_bf16_kernel<<<512, 256, 0, stream>>>(wsWg,  Wg,  512 * 512);
  cvt_bf16_kernel<<<768, 256, 0, stream>>>(wswih, wih, 768 * 512);
  cvt_bf16_kernel<<<384, 256, 0, stream>>>(wswhh, whh, 768 * 256);
  cvt_bf16_kernel<<<1024, 256, 0, stream>>>(wss2, seq2, B_ * L2_ * D2_);
  tcvt_bf16_kernel<<<256, 256, 0, stream>>>(wsW2t, W2, 512, 128);
  tcvt_bf16_kernel<<<128, 256, 0, stream>>>(wsW1t, W1, 256, 128);

  // seq2_proj = seq2 @ W1 : M = B*L2 = 2048 -> 128 M-tiles, 8 waves = 8 N-tiles
  seq2proj_kernel<<<(B_ * L2_) / 16, 256, 0, stream>>>(wsprj, wss2, wsW1t);

  // persistent recurrence: 2 WGs x 16 waves, 76KB dynamic LDS each
  const size_t smem = (size_t)(16 * H_ + 16 * P_ + 16 * L2_) * 4 +
                      (size_t)(3 * 16 * I_) * 2;  // 77824 B
  matchatt_persistent<<<B_ / 16, 512, smem, stream>>>(
      seq1, seq2, mask1, mask2, vattn, bg, bih, bhh,
      wsWg, wswih, wswhh, wsW2t, wsprj, out);
}
